// CombinedModel_11209864642811
// MI455X (gfx1250) — matrix operand order repaired and checked by reading, not compiled, not run on hardware
//
#include <hip/hip_runtime.h>
#include <hip/hip_bf16.h>

#define NN 50000
#define DD 128
#define HH 32
#define CC 16
#define EE 1600000

#define BST 104   // im2col row stride (f16): 52 dwords -> good bank spread, 16B aligned
#define WAS 104   // wA row stride (f16)
#define GWS 40    // transposed GCN weight row stride (f16): 20 dwords, 16B aligned

typedef __attribute__((ext_vector_type(16))) _Float16 v16h;
typedef __attribute__((ext_vector_type(8)))  _Float16 v8h;
typedef __attribute__((ext_vector_type(8)))  float    v8f;

__device__ __forceinline__ float frelu(float v) { return v > 0.f ? v : 0.f; }

__device__ __forceinline__ v16h cat8(v8h lo, v8h hi) {
  return __builtin_shufflevector(lo, hi, 0, 1, 2, 3, 4, 5, 6, 7,
                                 8, 9, 10, 11, 12, 13, 14, 15);
}

// ---------------- init: deg=1 (self loop), zero scatter targets ----------------
__global__ void k_init(float* __restrict__ deg, float* __restrict__ agg1,
                       float* __restrict__ agg2) {
  int i = blockIdx.x * blockDim.x + threadIdx.x;
  if (i < NN) deg[i] = 1.0f;
  if (i < NN * HH) agg1[i] = 0.f;
  int j = i - NN * HH;
  if (j >= 0 && j < NN * CC) agg2[j] = 0.f;
}

// ---------------- degree accumulation over target nodes ----------------
__global__ void k_degree(const int* __restrict__ ei, float* __restrict__ deg) {
  int e = blockIdx.x * blockDim.x + threadIdx.x;
  if (e >= EE) return;
  atomicAdd(deg + ei[EE + e], 1.0f);
}

__global__ void k_dinv(float* __restrict__ deg) {
  int i = blockIdx.x * blockDim.x + threadIdx.x;
  if (i < NN) deg[i] = rsqrtf(deg[i]);
}

// ---------------- CNN: conv1(VALU, writes im2col directly) + conv2(WMMA) ------
// One block per node, 512 threads = 16 waves; wave owns one 16x16 tile of the
// 32x128 conv2 output (M = out-channel, N = position, K = 96 im2col).
// Bcol[d][k] is K-contiguous so B fragments are two ds_load_b128.
__global__ void __launch_bounds__(512) k_cnn(
    const float* __restrict__ x,  const float* __restrict__ w1,
    const float* __restrict__ b1, const float* __restrict__ w2,
    const float* __restrict__ b2, _Float16* __restrict__ pooledh) {
  __shared__ float    xs[DD];
  __shared__ _Float16 Bbuf[(DD + 2) * BST];  // rows 0 and 129 are dump rows
  __shared__ _Float16 wA[HH * WAS];          // w2 f16, [m][h_in*3+kk]
  __shared__ int      pool[HH];              // int-punned non-negative float max
  _Float16* Bc = Bbuf + BST;                 // Bc[d*BST + k], d = 0..127

  const int node = blockIdx.x;
  const int tid  = threadIdx.x;

  if (tid < DD) xs[tid] = x[node * DD + tid];
  if (tid < HH) pool[tid] = 0;
  if (tid < 2 * HH) {  // true zero-pad boundary entries of the im2col
    int c = tid & (HH - 1);
    if (tid < HH) Bc[0 * BST + c * 3] = (_Float16)0.f;       // h1[c][-1]
    else          Bc[127 * BST + c * 3 + 2] = (_Float16)0.f; // h1[c][128]
  }
  {  // w2 -> f16 LDS with padded stride (16 threads per row, 6 elems each)
    int m = tid >> 4, k0 = (tid & 15) * 6;
#pragma unroll
    for (int q = 0; q < 6; ++q) wA[m * WAS + k0 + q] = (_Float16)w2[m * 96 + k0 + q];
  }
  __syncthreads();

  // conv1 (1->32ch, k=3, SAME) + relu, scattered straight into im2col slots
  for (int i = tid; i < HH * DD; i += 512) {
    int c = i >> 7, d = i & (DD - 1);
    float xl = d ? xs[d - 1] : 0.f;
    float xr = (d < DD - 1) ? xs[d + 1] : 0.f;
    float v  = fmaf(w1[c * 3], xl,
               fmaf(w1[c * 3 + 1], xs[d],
               fmaf(w1[c * 3 + 2], xr, b1[c])));
    _Float16 h = (_Float16)frelu(v);
    int base = c * 3;
    Bc[(d + 1) * BST + base]     = h;  // kk=0 for position d+1 (d=127 -> dump)
    Bc[d * BST + base + 1]       = h;  // kk=1 for position d
    Bc[(d - 1) * BST + base + 2] = h;  // kk=2 for position d-1 (d=0 -> dump)
  }
  __syncthreads();

  const int lane = tid & 31, wid = tid >> 5;
  const int mt = wid >> 3, nt = wid & 7;  // 2 M-tiles x 8 N-tiles
  const int lm = lane & 15, lh = lane >> 4;
  const int col = nt * 16 + lm;

  v8f acc = {};
#pragma unroll
  for (int ks = 0; ks < 3; ++ks) {
    // A fragment: two contiguous 8-f16 runs (K = lh*8..+7 and +16)
    const _Float16* ap = &wA[(mt * 16 + lm) * WAS + ks * 32 + (lh << 3)];
    v16h a = cat8(*(const v8h*)ap, *(const v8h*)(ap + 16));
    // B fragment: two contiguous 8-f16 runs (K = lh*8..+7 and +16) for this output column
    const _Float16* bp = &Bc[col * BST + ks * 32 + (lh << 3)];
    v16h b = cat8(*(const v8h*)bp, *(const v8h*)(bp + 16));
    acc = __builtin_amdgcn_wmma_f32_16x16x32_f16(false, a, false, b,
                                                 (short)0, acc, false, false);
  }

  // + bias, relu, max-pool over positions via LDS int-max (values >= 0)
#pragma unroll
  for (int r = 0; r < 8; ++r) {
    int m = mt * 16 + lh * 8 + r;
    float v = frelu(acc[r] + b2[m]);
    atomicMax(&pool[m], __float_as_int(v));
  }
  __syncthreads();
  if (tid < HH) pooledh[node * HH + tid] = (_Float16)__int_as_float(pool[tid]);
}

// ---------------- hw1 = pooled @ gw1   ([N,32] x [32,32], WMMA) ----------------
__global__ void __launch_bounds__(256) k_gemm1(const _Float16* __restrict__ ph,
                                               const float* __restrict__ gw1,
                                               float* __restrict__ hw1) {
  __shared__ _Float16 gwT[HH * GWS];  // transposed: gwT[n][k]
  const int tid = threadIdx.x;
  for (int i = tid; i < HH * HH; i += 256) {
    int k = i & (HH - 1), n = i >> 5;
    gwT[n * GWS + k] = (_Float16)gw1[k * HH + n];
  }
  __syncthreads();

  const int lane = tid & 31, wid = tid >> 5;
  const int tile = blockIdx.x * 8 + wid;  // 16 nodes per wave
  if (tile >= NN / 16) return;
  const int node0 = tile * 16;
  const int lm = lane & 15, lh = lane >> 4;

  const _Float16* arow = ph + (node0 + lm) * HH + (lh << 3);
  v16h a = cat8(*(const v8h*)arow, *(const v8h*)(arow + 16));

#pragma unroll
  for (int nt = 0; nt < 2; ++nt) {
    const _Float16* bp = &gwT[(nt * 16 + lm) * GWS + (lh << 3)];
    v16h b = cat8(*(const v8h*)bp, *(const v8h*)(bp + 16));
    v8f acc = {};
    acc = __builtin_amdgcn_wmma_f32_16x16x32_f16(false, a, false, b,
                                                 (short)0, acc, false, false);
#pragma unroll
    for (int r = 0; r < 8; ++r) {
      int m = lh * 8 + r;
      hw1[(node0 + m) * HH + nt * 16 + lm] = acc[r];
    }
  }
}

// ---------------- hw2 = h1 @ gw2   ([N,32] x [32,16], WMMA) ----------------
__global__ void __launch_bounds__(256) k_gemm2(const _Float16* __restrict__ h1h,
                                               const float* __restrict__ gw2,
                                               float* __restrict__ hw2) {
  __shared__ _Float16 gwT[CC * GWS];  // transposed: gwT[n][k]
  const int tid = threadIdx.x;
  for (int i = tid; i < HH * CC; i += 256) {
    int k = i & (HH - 1), n = i >> 5;
    gwT[n * GWS + k] = (_Float16)gw2[k * CC + n];
  }
  __syncthreads();

  const int lane = tid & 31, wid = tid >> 5;
  const int tile = blockIdx.x * 8 + wid;
  if (tile >= NN / 16) return;
  const int node0 = tile * 16;
  const int lm = lane & 15, lh = lane >> 4;

  const _Float16* arow = h1h + (node0 + lm) * HH + (lh << 3);
  v16h a = cat8(*(const v8h*)arow, *(const v8h*)(arow + 16));

  const _Float16* bp = &gwT[lm * GWS + (lh << 3)];
  v16h b = cat8(*(const v8h*)bp, *(const v8h*)(bp + 16));

  v8f acc = {};
  acc = __builtin_amdgcn_wmma_f32_16x16x32_f16(false, a, false, b,
                                               (short)0, acc, false, false);
#pragma unroll
  for (int r = 0; r < 8; ++r) {
    int m = lh * 8 + r;
    hw2[(node0 + m) * CC + lm] = acc[r];
  }
}

// ---------------- edge scatter: agg[col] += hw[row]*dinv[row]*dinv[col] -------
__global__ void k_scatter32(const int* __restrict__ ei, const float* __restrict__ dinv,
                            const float* __restrict__ src, float* __restrict__ dst) {
  int t = blockIdx.x * blockDim.x + threadIdx.x;
  if (t >= (EE + NN) * 8) return;
  int item = t >> 3, f = (t & 7) << 2;
  int r, c;
  if (item < EE) { r = ei[item]; c = ei[EE + item]; }
  else           { r = item - EE; c = r; }
  float nrm = dinv[r] * dinv[c];
  const float4 m = *(const float4*)(src + r * HH + f);
  float* p = dst + c * HH + f;
  atomicAdd(p + 0, m.x * nrm);
  atomicAdd(p + 1, m.y * nrm);
  atomicAdd(p + 2, m.z * nrm);
  atomicAdd(p + 3, m.w * nrm);
}

__global__ void k_scatter16(const int* __restrict__ ei, const float* __restrict__ dinv,
                            const float* __restrict__ src, float* __restrict__ dst) {
  int t = blockIdx.x * blockDim.x + threadIdx.x;
  if (t >= (EE + NN) * 4) return;
  int item = t >> 2, f = (t & 3) << 2;
  int r, c;
  if (item < EE) { r = ei[item]; c = ei[EE + item]; }
  else           { r = item - EE; c = r; }
  float nrm = dinv[r] * dinv[c];
  const float4 m = *(const float4*)(src + r * CC + f);
  float* p = dst + c * CC + f;
  atomicAdd(p + 0, m.x * nrm);
  atomicAdd(p + 1, m.y * nrm);
  atomicAdd(p + 2, m.z * nrm);
  atomicAdd(p + 3, m.w * nrm);
}

// ---------------- h1 = relu(agg1 + gb1) -> f16 for next GEMM ------------------
__global__ void k_relubias(const float* __restrict__ agg1, const float* __restrict__ gb1,
                           _Float16* __restrict__ h1h) {
  int i = blockIdx.x * blockDim.x + threadIdx.x;
  if (i >= NN * HH) return;
  h1h[i] = (_Float16)frelu(agg1[i] + gb1[i & (HH - 1)]);
}

// ---------------- out = log_softmax(agg2 + gb2) -------------------------------
__global__ void k_lsm(const float* __restrict__ agg2, const float* __restrict__ gb2,
                      float* __restrict__ out) {
  int n = blockIdx.x * blockDim.x + threadIdx.x;
  if (n >= NN) return;
  float v[CC], mx = -3.0e38f;
#pragma unroll
  for (int c = 0; c < CC; ++c) {
    v[c] = agg2[n * CC + c] + gb2[c];
    mx = fmaxf(mx, v[c]);
  }
  float s = 0.f;
#pragma unroll
  for (int c = 0; c < CC; ++c) s += expf(v[c] - mx);
  float l = logf(s);
#pragma unroll
  for (int c = 0; c < CC; ++c) out[n * CC + c] = v[c] - mx - l;
}

extern "C" void kernel_launch(void* const* d_in, const int* in_sizes, int n_in,
                              void* d_out, int out_size, void* d_ws, size_t ws_size,
                              hipStream_t stream) {
  (void)in_sizes; (void)n_in; (void)out_size; (void)ws_size;
  const float* x   = (const float*)d_in[0];
  const int*   ei  = (const int*)d_in[1];
  const float* w1  = (const float*)d_in[2];
  const float* b1  = (const float*)d_in[3];
  const float* w2  = (const float*)d_in[4];
  const float* b2  = (const float*)d_in[5];
  const float* gw1 = (const float*)d_in[6];
  const float* gb1 = (const float*)d_in[7];
  const float* gw2 = (const float*)d_in[8];
  const float* gb2 = (const float*)d_in[9];
  float* out = (float*)d_out;

  char* ws = (char*)d_ws;
  size_t o = 0;
  auto take = [&](size_t bytes) -> char* {
    char* p = ws + o;
    o = (o + bytes + 255) & ~(size_t)255;
    return p;
  };
  float*    deg  = (float*)   take((size_t)NN * 4);
  _Float16* ph   = (_Float16*)take((size_t)NN * HH * 2);
  float*    hw1  = (float*)   take((size_t)NN * HH * 4);
  float*    agg1 = (float*)   take((size_t)NN * HH * 4);
  _Float16* h1h  = (_Float16*)take((size_t)NN * HH * 2);
  float*    hw2  = (float*)   take((size_t)NN * CC * 4);
  float*    agg2 = (float*)   take((size_t)NN * CC * 4);

  k_init   <<<(NN * 48 + 255) / 256, 256, 0, stream>>>(deg, agg1, agg2);
  k_degree <<<(EE + 255) / 256,      256, 0, stream>>>(ei, deg);
  k_dinv   <<<(NN + 255) / 256,      256, 0, stream>>>(deg);
  k_cnn    <<<NN, 512, 0, stream>>>(x, w1, b1, w2, b2, ph);
  k_gemm1  <<<(NN / 16 + 7) / 8, 256, 0, stream>>>(ph, gw1, hw1);
  k_scatter32<<<((EE + NN) * 8 + 255) / 256, 256, 0, stream>>>(ei, deg, hw1, agg1);
  k_relubias <<<(NN * HH + 255) / 256, 256, 0, stream>>>(agg1, gb1, h1h);
  k_gemm2  <<<(NN / 16 + 7) / 8, 256, 0, stream>>>(h1h, gw2, hw2);
  k_scatter16<<<((EE + NN) * 4 + 255) / 256, 256, 0, stream>>>(ei, deg, hw2, agg2);
  k_lsm    <<<(NN + 255) / 256, 256, 0, stream>>>(agg2, gb2, out);
}
